// ParallelSpdUp_22067541967764
// MI455X (gfx1250) — matrix operand order repaired and checked by reading, not compiled, not run on hardware
//
#include <hip/hip_runtime.h>
#include <hip/hip_bf16.h>
#include <stdint.h>

// ---------------------------------------------------------------------------
// Problem: out = einsum("k,kod,bsd->bso", w, W[ids], x) + einsum("k,ko->o", w, b[ids])
// Folded:  Wmix = sum_k w[k]*W[ids[k]]  (bf16),  bmix = sum_k w[k]*b[ids[k]]
//          out[m,o] = sum_d xb[m,d]*Wmix[o,d] + bmix[o]   (one 4096x1024x1024 GEMM)
// GEMM uses v_wmma_f32_16x16x32_bf16 with a fully-unrolled, double-buffered
// global_load_async_to_lds_b128 pipeline (ASYNCcnt), one barrier per k-step.
// ---------------------------------------------------------------------------

typedef __attribute__((ext_vector_type(16))) __bf16 v16bf;
typedef __attribute__((ext_vector_type(8)))  float  v8f;

#define GM 4096      // rows of x / out   (B*S)
#define GN 1024      // output features
#define GK 1024      // input features
#define BM 128       // workgroup tile M
#define BN 128       // workgroup tile N
#define BK 32        // K step (one bf16 WMMA K)
#define NIT (GK / BK)
#define STR 40       // LDS row stride in halves (80B: 16B-aligned, bank-spread)

__device__ __forceinline__ unsigned short f32_to_bf16_rn(float f) {
  unsigned int u = __float_as_uint(f);
  unsigned int r = u + 0x7FFFu + ((u >> 16) & 1u);   // round-to-nearest-even
  return (unsigned short)(r >> 16);
}

// ---- 1) weighted mix of selected adapter matrices -> bf16 ------------------
__global__ __launch_bounds__(256) void mix_w_kernel(
    const float* __restrict__ W, const int* __restrict__ ids,
    const float* __restrict__ wts, int nk, unsigned short* __restrict__ Wm) {
  const int idx = blockIdx.x * 256 + threadIdx.x;          // 0 .. 1M-1
  float acc = 0.0f;
  for (int k = 0; k < nk; ++k) {
    acc += wts[k] * W[(size_t)ids[k] * (size_t)(GN * GK) + idx];
  }
  Wm[idx] = f32_to_bf16_rn(acc);
}

// ---- 2) x f32 -> bf16 (4 elements / thread) --------------------------------
__global__ __launch_bounds__(256) void cvt_x_kernel(
    const float* __restrict__ x, unsigned short* __restrict__ xb) {
  const int i = blockIdx.x * 256 + threadIdx.x;            // 0 .. 1M-1
  const float4 v = ((const float4*)x)[i];
  uint2 p;
  p.x = (unsigned int)f32_to_bf16_rn(v.x) | ((unsigned int)f32_to_bf16_rn(v.y) << 16);
  p.y = (unsigned int)f32_to_bf16_rn(v.z) | ((unsigned int)f32_to_bf16_rn(v.w) << 16);
  ((uint2*)xb)[i] = p;
}

// ---- 3) weighted mix of selected biases -> f32 -----------------------------
__global__ __launch_bounds__(256) void mix_b_kernel(
    const float* __restrict__ b, const int* __restrict__ ids,
    const float* __restrict__ wts, int nk, float* __restrict__ bias) {
  const int o = blockIdx.x * 256 + threadIdx.x;            // 0 .. 1023
  float acc = 0.0f;
  for (int k = 0; k < nk; ++k) acc += wts[k] * b[(size_t)ids[k] * GN + o];
  bias[o] = acc;
}

// ---- async global -> LDS: two b128 per call (32B). INST_OFFSET is added to
//      both the LDS and global addresses (ISA 10.7 / 15.18.3). --------------
__device__ __forceinline__ void async_copy_32B(uint32_t lds_addr,
                                               const unsigned short* g) {
  const uint64_t ga = (uint64_t)(uintptr_t)g;
  asm volatile("global_load_async_to_lds_b128 %0, %1, off"
               :: "v"(lds_addr), "v"(ga) : "memory");
  asm volatile("global_load_async_to_lds_b128 %0, %1, off offset:16"
               :: "v"(lds_addr), "v"(ga) : "memory");
}

// ---- 4) bf16 WMMA GEMM, double-buffered async-LDS pipeline -----------------
union FragA { uint4 q[2]; v16bf v; };

__global__ __launch_bounds__(256) void gemm_wmma_kernel(
    const unsigned short* __restrict__ Xb,   // [GM][GK] bf16
    const unsigned short* __restrict__ Wm,   // [GN][GK] bf16 (B operand: col o = row of Wm)
    const float* __restrict__ bias,          // [GN]
    float* __restrict__ out) {               // [GM][GN] f32
  __shared__ __align__(16) unsigned short Xs[2][BM * STR];
  __shared__ __align__(16) unsigned short Bs[2][BN * STR];

  const int tid   = threadIdx.x;
  const int lane  = tid & 31;
  const int wave  = tid >> 5;        // 0..7
  const int waveM = wave >> 2;       // 0..1  -> 64-row slab
  const int waveN = wave & 3;        // 0..3  -> 32-col slab
  const int mBase = blockIdx.y * BM;
  const int nBase = blockIdx.x * BN;

  const int lh = lane & 15;          // fragment row/col within 16
  const int hi = lane >> 4;          // K half-select per ISA layout

  v8f c[4][2];
#pragma unroll
  for (int mt = 0; mt < 4; ++mt)
#pragma unroll
    for (int nt = 0; nt < 2; ++nt) c[mt][nt] = (v8f){};

  // cooperative staging assignment: 256 threads x 32B = 128 rows x 64B
  const int srow = tid >> 1;         // 0..127
  const int sseg = tid & 1;          // which 16-half (32B) segment
  const unsigned short* gX = Xb + (size_t)(mBase + srow) * GK + sseg * 16;
  const unsigned short* gW = Wm + (size_t)(nBase + srow) * GK + sseg * 16;
  const int sOff = srow * STR + sseg * 16;

  // prologue: stage tile 0 into buffer 0
  async_copy_32B((uint32_t)(uintptr_t)&Xs[0][sOff], gX);
  async_copy_32B((uint32_t)(uintptr_t)&Bs[0][sOff], gW);

#pragma unroll
  for (int it = 0; it < NIT; ++it) {
    const int cur = it & 1;          // compile-time constant after full unroll
    const int k0  = it * BK;
    // my async copies for tile `it` have landed in LDS:
    asm volatile("s_wait_asynccnt 0x0" ::: "memory");
    // everyone's copies landed + everyone finished reading buffer cur^1:
    __syncthreads();
    // kick off tile it+1 into the other buffer; overlaps with WMMAs below
    if (it + 1 < NIT) {
      async_copy_32B((uint32_t)(uintptr_t)&Xs[cur ^ 1][sOff], gX + (k0 + BK));
      async_copy_32B((uint32_t)(uintptr_t)&Bs[cur ^ 1][sOff], gW + (k0 + BK));
    }

    // ---- B fragments: lane lh = column o, halves 0..7 = K hi*8.., halves 8..15 = K 16+hi*8.. ----
    FragA bf[2];
#pragma unroll
    for (int nt = 0; nt < 2; ++nt) {
      const int o = waveN * 32 + nt * 16 + lh;
      const uint4* p = (const uint4*)(&Bs[cur][o * STR + hi * 8]);
      bf[nt].q[0] = p[0];
      bf[nt].q[1] = p[2];
    }

    // ---- A fragments: lane lh = row m; halves 0..7 = K hi*8..,
    //      halves 8..15 = K 16+hi*8.. (ISA 16-bit A layout) ----
#pragma unroll
    for (int mt = 0; mt < 4; ++mt) {
      const int r = waveM * 64 + mt * 16 + lh;
      FragA af;
      af.q[0] = *(const uint4*)(&Xs[cur][r * STR + hi * 8]);
      af.q[1] = *(const uint4*)(&Xs[cur][r * STR + 16 + hi * 8]);
#pragma unroll
      for (int nt = 0; nt < 2; ++nt) {
        c[mt][nt] = __builtin_amdgcn_wmma_f32_16x16x32_bf16(
            /*neg_a=*/false, af.v, /*neg_b=*/false, bf[nt].v,
            /*c_mod=*/(short)0, c[mt][nt], /*reuse_a=*/false, /*reuse_b=*/false);
      }
    }
    // no trailing barrier: next iteration's top barrier orders buffer reuse
  }

  // ---- epilogue: C layout VGPR v -> M = v + hi*8, N = lh ----
#pragma unroll
  for (int nt = 0; nt < 2; ++nt) {
    const int o = nBase + waveN * 32 + nt * 16 + lh;
    const float bo = bias[o];
#pragma unroll
    for (int mt = 0; mt < 4; ++mt) {
      const int mb = mBase + waveM * 64 + mt * 16 + hi * 8;
#pragma unroll
      for (int v = 0; v < 8; ++v) {
        out[(size_t)(mb + v) * GN + o] = c[mt][nt][v] + bo;
      }
    }
  }
}

// ---------------------------------------------------------------------------
extern "C" void kernel_launch(void* const* d_in, const int* in_sizes, int n_in,
                              void* d_out, int out_size, void* d_ws, size_t ws_size,
                              hipStream_t stream) {
  const float* x   = (const float*)d_in[0];   // [2,2048,1024]
  const float* W   = (const float*)d_in[1];   // [32,1024,1024]
  const float* b   = (const float*)d_in[2];   // [32,1024]
  const int*   ids = (const int*)d_in[3];     // [K]
  const float* wts = (const float*)d_in[4];   // [K]
  const int nk = in_sizes[3];
  float* out = (float*)d_out;                 // [2,2048,1024]

  // workspace carve-up (~10 MB)
  unsigned short* xb   = (unsigned short*)d_ws;                              // 8 MB
  unsigned short* wm   = (unsigned short*)((char*)d_ws + ((size_t)8 << 20)); // 2 MB
  float*          bias = (float*)((char*)d_ws + ((size_t)10 << 20));         // 4 KB
  (void)ws_size; (void)n_in; (void)out_size;

  mix_w_kernel<<<dim3((GN * GK) / 256), dim3(256), 0, stream>>>(W, ids, wts, nk, wm);
  cvt_x_kernel<<<dim3((GM * GK) / (4 * 256)), dim3(256), 0, stream>>>(x, xb);
  mix_b_kernel<<<dim3(GN / 256), dim3(256), 0, stream>>>(b, ids, wts, nk, bias);
  gemm_wmma_kernel<<<dim3(GN / BN, GM / BM), dim3(256), 0, stream>>>(xb, wm, bias, out);
}
